// ResidualFNO_5239860101554
// MI455X (gfx1250) — matrix operand order repaired and checked by reading, not compiled, not run on hardware
//
#include <hip/hip_runtime.h>
#include <hip/hip_bf16.h>

typedef __attribute__((ext_vector_type(16))) _Float16 v16h;
typedef __attribute__((ext_vector_type(8)))  float    v8f;

#define NB 2097152
#define NLAY 4
#define EPSV 1e-5f

// silu via v_exp_f32 + v_rcp_f32 (avoids IEEE div_fixup sequence)
__device__ __forceinline__ float silu_f(float x) {
    return x * __builtin_amdgcn_rcpf(1.0f + __expf(-x));
}

// ---------------------------------------------------------------------------
// Zero the stats accumulators (must run every launch; ws is not re-poisoned).
// ---------------------------------------------------------------------------
__global__ void zero_kernel(float* __restrict__ p, int n) {
    int i = blockIdx.x * blockDim.x + threadIdx.x;
    if (i < n) p[i] = 0.0f;
}

// ---------------------------------------------------------------------------
// Pack kernel (1 workgroup): build f16 B-matrix WMMA fragments for
// lin1_w[l], lin2_w[l], and the folded Fourier matrix F_l.
// B layout (16-bit, 32x16 KxN): lanes0-15 hold k=0..7 (e0..7) & 16..23 (e8..15),
// lanes16-31 hold k=8..15 & 24..31; n = lane & 15.  => k = (e<8?0:8)+(lane>=16?8:0)+e.
// Fragment storage: frag[tile*512 + lane*16 + e], tile selects n-half.
// Per layer: [0,1024) lin1, [1024,2048) lin2, [2048,3072) F.
// ---------------------------------------------------------------------------
__global__ __launch_bounds__(256)
void pack_kernel(const float* __restrict__ fno_wr, const float* __restrict__ fno_wi,
                 const float* __restrict__ lin1_w, const float* __restrict__ lin2_w,
                 _Float16* __restrict__ frag) {
    __shared__ float F[1024];
    const int tid = threadIdx.x;
    const float c0 = 0.19634954084936207f; // 2*pi/32
    for (int l = 0; l < NLAY; ++l) {
        _Float16* fl = frag + l * 3072;
        // lin1 / lin2 fragments straight from global ([in,out] layout => W[k][n])
        for (int idx = tid; idx < 1024; idx += 256) {
            int tile = idx >> 9, lane = (idx >> 4) & 31, e = idx & 15;
            int k = ((e < 8) ? 0 : 8) + ((lane >= 16) ? 8 : 0) + e;
            int n = tile * 16 + (lane & 15);
            fl[idx]        = (_Float16)lin1_w[l * 1024 + k * 32 + n];
            fl[1024 + idx] = (_Float16)lin2_w[l * 1024 + k * 32 + n];
        }
        // Fold rfft -> complex mix -> irfft into real 32x32 matrix F:
        // F[m][n] = (1/32) sum_{j,k} w_j (Wr[k][j] cos th - Wi[k][j] sin th),
        // th = 2pi(n*j - m*k)/32, w_0 = w_16 = 1, else 2.
        const float* wr = fno_wr + l * 289;
        const float* wi = fno_wi + l * 289;
        for (int idx = tid; idx < 1024; idx += 256) {
            int m = idx >> 5, n = idx & 31;
            float acc = 0.0f;
            for (int j = 0; j <= 16; ++j) {
                float wj = (j == 0 || j == 16) ? 1.0f : 2.0f;
                for (int k = 0; k <= 16; ++k) {
                    float th = c0 * (float)(n * j - m * k);
                    acc += wj * (wr[k * 17 + j] * cosf(th) - wi[k * 17 + j] * sinf(th));
                }
            }
            F[idx] = acc * (1.0f / 32.0f);
        }
        __syncthreads();
        for (int idx = tid; idx < 1024; idx += 256) {
            int tile = idx >> 9, lane = (idx >> 4) & 31, e = idx & 15;
            int k = ((e < 8) ? 0 : 8) + ((lane >= 16) ? 8 : 0) + e;
            int n = tile * 16 + (lane & 15);
            fl[2048 + idx] = (_Float16)F[k * 32 + n];
        }
        __syncthreads();
    }
}

// ---------------------------------------------------------------------------
// Stem: h = x @ stem_w + stem_b, plus sum/sumsq of h (bn1 stats for layer 0).
// 256 thr * 8 rows/thr -> 2048 rows/WG, grid 1024.
// ---------------------------------------------------------------------------
__global__ __launch_bounds__(256)
void stem_kernel(const float* __restrict__ x, const float* __restrict__ w,
                 const float* __restrict__ b, float* __restrict__ h,
                 float* __restrict__ sOut) {
    __shared__ float W[320];
    __shared__ float Bv[32];
    __shared__ float lds[64];
    const int tid = threadIdx.x;
    if (tid < 320) W[tid] = w[tid];
    if (tid < 32)  Bv[tid] = b[tid];
    if (tid < 64)  lds[tid] = 0.0f;
    __syncthreads();
    float s[32], q[32];
    #pragma unroll
    for (int c = 0; c < 32; ++c) { s[c] = 0.0f; q[c] = 0.0f; }
    const size_t base = ((size_t)blockIdx.x * 256 + tid) * 8;
    for (int rr = 0; rr < 8; ++rr) {
        const float* xp = x + (base + rr) * 10;
        float xi[10];
        #pragma unroll
        for (int i = 0; i < 10; ++i) xi[i] = xp[i];
        float* hp = h + (base + rr) * 32;
        #pragma unroll
        for (int c = 0; c < 32; ++c) {
            float acc = Bv[c];
            #pragma unroll
            for (int i = 0; i < 10; ++i) acc += xi[i] * W[i * 32 + c];
            hp[c] = acc;
            s[c] += acc; q[c] += acc * acc;
        }
    }
    #pragma unroll
    for (int c = 0; c < 32; ++c) {
        atomicAdd(&lds[c], s[c]);
        atomicAdd(&lds[32 + c], q[c]);
    }
    __syncthreads();
    if (tid < 64) atomicAdd(&sOut[tid], lds[tid]);
}

// ---------------------------------------------------------------------------
// K1: t = silu(bn1(h)) @ lin1_w + lin1_b ; accumulate t stats (for bn2).
// Wave owns 4 tiles of 16 rows; 8 waves -> 512 rows/WG, grid 4096.
// A layout (16-bit 16x32): lane<16: m=lane, K=0..7 (e0..7) & 16..23 (e8..15);
// lane>=16: m=lane-16, K=8..15 & 24..31.
// ---------------------------------------------------------------------------
__global__ __launch_bounds__(256)
void k1_kernel(const float* __restrict__ h, float* __restrict__ t,
               const _Float16* __restrict__ fragL1,
               const float* __restrict__ g, const float* __restrict__ bb,
               const float* __restrict__ bias,
               const float* __restrict__ sIn, float* __restrict__ sOut) {
    __shared__ float lds[64];
    const int tid = threadIdx.x, wave = tid >> 5, lane = tid & 31;
    const float invB = 1.0f / (float)NB;
    const int kb = (lane >= 16) ? 8 : 0;
    const int nn = lane & 15;
    float sc[16], sh[16];
    #pragma unroll
    for (int i = 0; i < 16; ++i) {
        int ch = (i < 8) ? (kb + i) : (16 + kb + (i - 8));
        float mean = sIn[ch] * invB;
        float var  = sIn[32 + ch] * invB - mean * mean;
        float s = g[ch] * rsqrtf(var + EPSV);
        sc[i] = s; sh[i] = bb[ch] - mean * s;
    }
    v16h b0 = *(const v16h*)(fragL1 + lane * 16);
    v16h b1 = *(const v16h*)(fragL1 + 512 + lane * 16);
    const float bias0 = bias[nn], bias1 = bias[16 + nn];
    float s0 = 0, q0 = 0, s1 = 0, q1 = 0;
    const size_t base = (size_t)blockIdx.x * 512 + (size_t)wave * 64;
    #pragma unroll
    for (int tt = 0; tt < 4; ++tt) {
        const float* hr = h + (base + tt * 16 + nn) * 32;
        // explicit b128 loads: 4x float4 per lane
        const float4* p0 = (const float4*)(hr + kb);
        const float4* p1 = (const float4*)(hr + 16 + kb);
        float4 hA = p0[0], hB = p0[1], hC = p1[0], hD = p1[1];
        float hv[16] = { hA.x, hA.y, hA.z, hA.w, hB.x, hB.y, hB.z, hB.w,
                         hC.x, hC.y, hC.z, hC.w, hD.x, hD.y, hD.z, hD.w };
        v16h a;
        #pragma unroll
        for (int i = 0; i < 16; ++i) {
            float xv = hv[i] * sc[i] + sh[i];
            a[i] = (_Float16)silu_f(xv);
        }
        v8f c0, c1;
        #pragma unroll
        for (int r = 0; r < 8; ++r) { c0[r] = bias0; c1[r] = bias1; }
        c0 = __builtin_amdgcn_wmma_f32_16x16x32_f16(false, a, false, b0, (short)0, c0, false, false);
        c1 = __builtin_amdgcn_wmma_f32_16x16x32_f16(false, a, false, b1, (short)0, c1, false, false);
        float* tp = t + (base + tt * 16 + ((lane >= 16) ? 8 : 0)) * 32;
        #pragma unroll
        for (int r = 0; r < 8; ++r) {
            float v0 = c0[r], v1 = c1[r];
            tp[r * 32 + nn] = v0;
            tp[r * 32 + 16 + nn] = v1;
            s0 += v0; q0 += v0 * v0;
            s1 += v1; q1 += v1 * v1;
        }
    }
    if (tid < 64) lds[tid] = 0.0f;
    __syncthreads();
    atomicAdd(&lds[nn], s0);       atomicAdd(&lds[32 + nn], q0);
    atomicAdd(&lds[16 + nn], s1);  atomicAdd(&lds[48 + nn], q1);
    __syncthreads();
    if (tid < 64) atomicAdd(&sOut[tid], lds[tid]);
}

// ---------------------------------------------------------------------------
// K2: h = silu( h@F  +  silu(bn2(t))@lin2_w + lin2_b  +  h ),
// accumulate new-h stats (bn1 of next layer). In-place on h (each wave's
// loads precede its stores in program order; rows are wave-private).
// ---------------------------------------------------------------------------
__global__ __launch_bounds__(256)
void k2_kernel(float* __restrict__ h, const float* __restrict__ t,
               const _Float16* __restrict__ fragL2, const _Float16* __restrict__ fragF,
               const float* __restrict__ g, const float* __restrict__ bb,
               const float* __restrict__ bias,
               const float* __restrict__ sIn, float* __restrict__ sOut) {
    __shared__ float lds[64];
    const int tid = threadIdx.x, wave = tid >> 5, lane = tid & 31;
    const float invB = 1.0f / (float)NB;
    const int kb = (lane >= 16) ? 8 : 0;
    const int nn = lane & 15;
    float sc[16], sh[16];
    #pragma unroll
    for (int i = 0; i < 16; ++i) {
        int ch = (i < 8) ? (kb + i) : (16 + kb + (i - 8));
        float mean = sIn[ch] * invB;
        float var  = sIn[32 + ch] * invB - mean * mean;
        float s = g[ch] * rsqrtf(var + EPSV);
        sc[i] = s; sh[i] = bb[ch] - mean * s;
    }
    v16h bl0 = *(const v16h*)(fragL2 + lane * 16);
    v16h bl1 = *(const v16h*)(fragL2 + 512 + lane * 16);
    v16h bf0 = *(const v16h*)(fragF + lane * 16);
    v16h bf1 = *(const v16h*)(fragF + 512 + lane * 16);
    const float bias0 = bias[nn], bias1 = bias[16 + nn];
    float s0 = 0, q0 = 0, s1 = 0, q1 = 0;
    const size_t base = (size_t)blockIdx.x * 512 + (size_t)wave * 64;
    #pragma unroll
    for (int tt = 0; tt < 4; ++tt) {
        const float* tr = t + (base + tt * 16 + nn) * 32;
        const float* hr = h + (base + tt * 16 + nn) * 32;
        const float4* tp0 = (const float4*)(tr + kb);
        const float4* tp1 = (const float4*)(tr + 16 + kb);
        const float4* hp0 = (const float4*)(hr + kb);
        const float4* hp1 = (const float4*)(hr + 16 + kb);
        float4 tA = tp0[0], tB = tp0[1], tC = tp1[0], tD = tp1[1];
        float4 gA = hp0[0], gB = hp0[1], gC = hp1[0], gD = hp1[1];
        float tv[16] = { tA.x, tA.y, tA.z, tA.w, tB.x, tB.y, tB.z, tB.w,
                         tC.x, tC.y, tC.z, tC.w, tD.x, tD.y, tD.z, tD.w };
        float gv[16] = { gA.x, gA.y, gA.z, gA.w, gB.x, gB.y, gB.z, gB.w,
                         gC.x, gC.y, gC.z, gC.w, gD.x, gD.y, gD.z, gD.w };
        v16h au, ah;
        #pragma unroll
        for (int i = 0; i < 16; ++i) {
            float uv = tv[i] * sc[i] + sh[i];
            au[i] = (_Float16)silu_f(uv);
            ah[i] = (_Float16)gv[i];
        }
        v8f r0, r1, x0, x1;
        #pragma unroll
        for (int r = 0; r < 8; ++r) { r0[r] = bias0; r1[r] = bias1; x0[r] = 0.0f; x1[r] = 0.0f; }
        r0 = __builtin_amdgcn_wmma_f32_16x16x32_f16(false, au, false, bl0, (short)0, r0, false, false);
        r1 = __builtin_amdgcn_wmma_f32_16x16x32_f16(false, au, false, bl1, (short)0, r1, false, false);
        x0 = __builtin_amdgcn_wmma_f32_16x16x32_f16(false, ah, false, bf0, (short)0, x0, false, false);
        x1 = __builtin_amdgcn_wmma_f32_16x16x32_f16(false, ah, false, bf1, (short)0, x1, false, false);
        float* hp = h + (base + tt * 16 + ((lane >= 16) ? 8 : 0)) * 32;
        #pragma unroll
        for (int r = 0; r < 8; ++r) {
            float hv0 = hp[r * 32 + nn];
            float hv1 = hp[r * 32 + 16 + nn];
            float v0 = silu_f(x0[r] + r0[r] + hv0);
            float v1 = silu_f(x1[r] + r1[r] + hv1);
            hp[r * 32 + nn] = v0;
            hp[r * 32 + 16 + nn] = v1;
            s0 += v0; q0 += v0 * v0;
            s1 += v1; q1 += v1 * v1;
        }
    }
    if (tid < 64) lds[tid] = 0.0f;
    __syncthreads();
    atomicAdd(&lds[nn], s0);       atomicAdd(&lds[32 + nn], q0);
    atomicAdd(&lds[16 + nn], s1);  atomicAdd(&lds[48 + nn], q1);
    __syncthreads();
    if (tid < 64) atomicAdd(&sOut[tid], lds[tid]);
}

// ---------------------------------------------------------------------------
// Head: out = h @ head_w + head_b.  256 thr * 4 rows -> grid 2048.
// ---------------------------------------------------------------------------
__global__ __launch_bounds__(256)
void head_kernel(const float* __restrict__ h, const float* __restrict__ w,
                 const float* __restrict__ b, float* __restrict__ out) {
    __shared__ float W[320];
    __shared__ float Bv[10];
    const int tid = threadIdx.x;
    if (tid < 320) W[tid] = w[tid];
    if (tid < 10)  Bv[tid] = b[tid];
    __syncthreads();
    const size_t base = ((size_t)blockIdx.x * 256 + tid) * 4;
    for (int rr = 0; rr < 4; ++rr) {
        const float4* hp = (const float4*)(h + (base + rr) * 32);
        float4 v0 = hp[0], v1 = hp[1], v2 = hp[2], v3 = hp[3];
        float4 v4 = hp[4], v5 = hp[5], v6 = hp[6], v7 = hp[7];
        float hv[32] = { v0.x,v0.y,v0.z,v0.w, v1.x,v1.y,v1.z,v1.w,
                         v2.x,v2.y,v2.z,v2.w, v3.x,v3.y,v3.z,v3.w,
                         v4.x,v4.y,v4.z,v4.w, v5.x,v5.y,v5.z,v5.w,
                         v6.x,v6.y,v6.z,v6.w, v7.x,v7.y,v7.z,v7.w };
        float* op = out + (base + rr) * 10;
        #pragma unroll
        for (int j = 0; j < 10; ++j) {
            float acc = Bv[j];
            #pragma unroll
            for (int c = 0; c < 32; ++c) acc += hv[c] * W[c * 10 + j];
            op[j] = acc;
        }
    }
}

extern "C" void kernel_launch(void* const* d_in, const int* in_sizes, int n_in,
                              void* d_out, int out_size, void* d_ws, size_t ws_size,
                              hipStream_t stream) {
    (void)in_sizes; (void)n_in; (void)out_size; (void)ws_size;
    const float* x      = (const float*)d_in[0];
    const float* stem_w = (const float*)d_in[1];
    const float* stem_b = (const float*)d_in[2];
    const float* fno_wr = (const float*)d_in[3];
    const float* fno_wi = (const float*)d_in[4];
    const float* bn1_g  = (const float*)d_in[5];
    const float* bn1_b  = (const float*)d_in[6];
    const float* lin1_w = (const float*)d_in[7];
    const float* lin1_b = (const float*)d_in[8];
    const float* bn2_g  = (const float*)d_in[9];
    const float* bn2_b  = (const float*)d_in[10];
    const float* lin2_w = (const float*)d_in[11];
    const float* lin2_b = (const float*)d_in[12];
    const float* head_w = (const float*)d_in[13];
    const float* head_b = (const float*)d_in[14];

    // Workspace layout: h[B*32] f32 | t[B*32] f32 | stats (9 stages * 64) | frags
    float* h = (float*)d_ws;
    const size_t nh = (size_t)NB * 32;
    float* t = h + nh;
    float* stats = t + nh;                      // 576 floats used, pad to 1024
    _Float16* frag = (_Float16*)(stats + 1024); // 4 layers * 3072 f16

    zero_kernel<<<3, 256, 0, stream>>>(stats, 576);
    pack_kernel<<<1, 256, 0, stream>>>(fno_wr, fno_wi, lin1_w, lin2_w, frag);
    stem_kernel<<<NB / 2048, 256, 0, stream>>>(x, stem_w, stem_b, h, stats);
    for (int l = 0; l < NLAY; ++l) {
        k1_kernel<<<NB / 512, 256, 0, stream>>>(
            h, t, frag + (size_t)l * 3072,
            bn1_g + l * 32, bn1_b + l * 32, lin1_b + l * 32,
            stats + (2 * l) * 64, stats + (2 * l + 1) * 64);
        k2_kernel<<<NB / 512, 256, 0, stream>>>(
            h, t, frag + (size_t)l * 3072 + 1024, frag + (size_t)l * 3072 + 2048,
            bn2_g + l * 32, bn2_b + l * 32, lin2_b + l * 32,
            stats + (2 * l + 1) * 64, stats + (2 * l + 2) * 64);
    }
    head_kernel<<<NB / 1024, 256, 0, stream>>>(h, head_w, head_b, (float*)d_out);
}
